// FlaxGrok1SparseMoeBlock_62629213110304
// MI455X (gfx1250) — matrix operand order repaired and checked by reading, not compiled, not run on hardware
//
#include <hip/hip_runtime.h>
#include <math.h>
#include <stdint.h>

// Problem constants (from reference)
#define E_ 8
#define H_ 1024
#define I_ 4096
#define T_ 2048   // B*S = 2*1024

typedef __bf16 bf16_t;
typedef bf16_t v16bf __attribute__((ext_vector_type(16)));
typedef bf16_t v8bf  __attribute__((ext_vector_type(8)));
typedef bf16_t v4bf  __attribute__((ext_vector_type(4)));
typedef float  v8f_t __attribute__((ext_vector_type(8)));

typedef unsigned int u32x4 __attribute__((ext_vector_type(4)));
typedef int          i32x4 __attribute__((ext_vector_type(4)));
typedef int          i32x8 __attribute__((ext_vector_type(8)));

union V16u { v16bf v; v8bf h[2]; };

#define PAD 40  // LDS row stride in bf16 (80B = 20 DWORDs: 16B-aligned, bank-skewed)

__device__ __forceinline__ float gelu_tanh(float x) {
  const float c0 = 0.7978845608028654f;
  const float c1 = 0.044715f;
  return 0.5f * x * (1.0f + tanhf(c0 * (x + c1 * x * x * x)));
}

// A-frag (16x32 bf16): lane<16 -> K{0..7,16..23}; lane>=16 -> K{8..15,24..31}.
__device__ __forceinline__ v16bf ldsA(const bf16_t* row, int klo) {
  V16u u;
  u.h[0] = *(const v8bf*)(row + klo);
  u.h[1] = *(const v8bf*)(row + klo + 16);
  return u.v;
}
// B-frag (32x16 bf16): lane<16 -> K{0..7,16..23}; lane>=16 -> K{8..15,24..31}.
__device__ __forceinline__ v16bf ldsB(const bf16_t* p) {
  V16u u;
  u.h[0] = *(const v8bf*)(p);
  u.h[1] = *(const v8bf*)(p + 16);
  return u.v;
}

__device__ __forceinline__ v8f_t wmma_bf16(v16bf a, v16bf b, v8f_t c) {
  return __builtin_amdgcn_wmma_f32_16x16x32_bf16(
      /*neg_a=*/false, a, /*neg_b=*/false, b,
      /*c_mod=*/(short)0, c, /*reuse_a=*/false, /*reuse_b=*/false);
}

// ----------------------------------------------------------------- TDM
// Issue a 2D tensor_load_to_lds: tile_rows x tile_elems_x bf16 elements,
// row-major source with row_stride_elems, LDS destination padded by 4 DWORDs
// every 16 DWORDs (row = 32 bf16 = 16 DWORDs -> padded stride = PAD bf16).
// D# layout per CDNA5 ISA ch.8 (group0 128b, group1 256b).
__device__ __forceinline__ void tdm_load_2d_bf16(
    unsigned lds_off, const bf16_t* gaddr,
    unsigned tile_elems_x, unsigned tile_rows,
    unsigned tensor_dim0, unsigned tensor_dim1, unsigned row_stride_elems) {
  unsigned long long ga = (unsigned long long)(uintptr_t)gaddr;
  u32x4 g0;
  g0.x = 1u;                                  // count=1, user descriptor
  g0.y = lds_off;                             // lds_addr (bytes)
  g0.z = (unsigned)(ga & 0xFFFFFFFFull);      // global_addr[31:0]
  g0.w = (unsigned)((ga >> 32) & 0x01FFFFFFull) | (2u << 30);  // addr[56:32]|type=2
  i32x8 g1;
  // data_size=1 (2B), pad_enable=1, pad_interval=3 (16 DW), pad_amount=3 (4 DW)
  g1[0] = (int)((1u << 16) | (1u << 20) | (3u << 22) | (3u << 25));
  g1[1] = (int)((tensor_dim0 & 0xFFFFu) << 16);              // td0 lo16
  g1[2] = (int)(((tensor_dim0 >> 16) & 0xFFFFu) |
                ((tensor_dim1 & 0xFFFFu) << 16));            // td0 hi | td1 lo
  g1[3] = (int)(((tensor_dim1 >> 16) & 0xFFFFu) |
                ((tile_elems_x & 0xFFFFu) << 16));           // td1 hi | tile_dim0
  g1[4] = (int)(tile_rows & 0xFFFFu);                        // tile_dim1; dim2=0
  g1[5] = (int)row_stride_elems;                             // stride0[31:0]
  g1[6] = 0;                                                 // stride0 hi | stride1 lo
  g1[7] = 0;                                                 // stride1 hi
  i32x4 gz = {0, 0, 0, 0};
#if __clang_major__ >= 23
  i32x8 gz8 = {0, 0, 0, 0, 0, 0, 0, 0};
  __builtin_amdgcn_tensor_load_to_lds(g0, g1, gz, gz, gz8, 0);
#else
  __builtin_amdgcn_tensor_load_to_lds(g0, g1, gz, gz, 0);
#endif
}

// ---------------------------------------------------------------- converts
__global__ void cvt_f32_bf16(const float* __restrict__ src,
                             bf16_t* __restrict__ dst, size_t n4) {
  size_t i = (size_t)blockIdx.x * blockDim.x + threadIdx.x;
  size_t stride = (size_t)gridDim.x * blockDim.x;
  for (; i < n4; i += stride) {
    float4 f = ((const float4*)src)[i];
    v4bf o;
    o.x = (bf16_t)f.x; o.y = (bf16_t)f.y;
    o.z = (bf16_t)f.z; o.w = (bf16_t)f.w;
    ((v4bf*)dst)[i] = o;
  }
}

// ---------------------------------------------------------------- router
__global__ void router_kernel(const float* __restrict__ x,
                              const float* __restrict__ Wg,
                              float* __restrict__ logits,
                              float* __restrict__ comb) {
  int wid = threadIdx.x >> 5;
  int lane = threadIdx.x & 31;
  int t = blockIdx.x * 8 + wid;
  if (t >= T_) return;
  const float* xr = x + (size_t)t * H_;
  float acc[E_];
#pragma unroll
  for (int e = 0; e < E_; ++e) acc[e] = 0.f;
  for (int h = lane; h < H_; h += 32) {
    float xv = xr[h];
    const float* wr = Wg + (size_t)h * E_;
#pragma unroll
    for (int e = 0; e < E_; ++e) acc[e] += xv * wr[e];
  }
#pragma unroll
  for (int e = 0; e < E_; ++e) {
#pragma unroll
    for (int off = 16; off > 0; off >>= 1)
      acc[e] += __shfl_xor(acc[e], off, 32);
  }
  if (lane == 0) {
    int i1 = 0;
#pragma unroll
    for (int e = 1; e < E_; ++e) if (acc[e] > acc[i1]) i1 = e;
    int i2 = (i1 == 0) ? 1 : 0;
#pragma unroll
    for (int e = 0; e < E_; ++e)
      if (e != i1 && acc[e] > acc[i2]) i2 = e;
    float p = 1.f / (1.f + __expf(acc[i2] - acc[i1]));  // softmax over top-2
#pragma unroll
    for (int e = 0; e < E_; ++e) {
      logits[(size_t)t * E_ + e] = acc[e];
      comb[t * E_ + e] = (e == i1) ? p : ((e == i2) ? (1.f - p) : 0.f);
    }
  }
}

// --------------------------------------------------- gate/value fused GEMM
// Block tile: 128 (tokens) x 64 (I cols), K over H. X tile arrives via TDM
// (wave 0 issues, s_wait_tensorcnt before the barrier); W tiles go through
// registers with an LDS-transpose store. 8 waves x 32x32, dual accumulators.
__global__ void __launch_bounds__(256)
gemm_gluact(const bf16_t* __restrict__ Xb, const bf16_t* __restrict__ Winb,
            const bf16_t* __restrict__ Wvb, const float* __restrict__ comb,
            bf16_t* __restrict__ Ap) {
  __shared__ __align__(16) bf16_t Xs[2][128][PAD];
  __shared__ __align__(16) bf16_t Gs[2][64][PAD];   // Win tile, [n][k]
  __shared__ __align__(16) bf16_t Vs[2][64][PAD];   // Wv  tile, [n][k]

  const int e  = blockIdx.z;
  const int M0 = blockIdx.y * 128;
  const int N0 = blockIdx.x * 64;
  const bf16_t* WinE = Winb + (size_t)e * H_ * I_;
  const bf16_t* WvE  = Wvb  + (size_t)e * H_ * I_;

  const int tid  = threadIdx.x;
  const int wid  = tid >> 5, lane = tid & 31;
  const int wm   = wid >> 1, wn = wid & 1;
  const int mrow = wm * 32, ncol = wn * 32;
  const int lr   = lane & 15;
  const int klo  = (lane < 16) ? 0 : 8;
  const int kb   = (lane < 16) ? 0 : 8;

  const int u  = tid & 127;  // W chunk-pair id
  const bf16_t* Wsrc = (tid < 128) ? WinE : WvE;

  const v8f_t zf = {0.f, 0.f, 0.f, 0.f, 0.f, 0.f, 0.f, 0.f};
  v8f_t accg[2][2], accv[2][2];
#pragma unroll
  for (int i = 0; i < 2; ++i)
#pragma unroll
    for (int j = 0; j < 2; ++j) { accg[i][j] = zf; accv[i][j] = zf; }

  uint4 wr0, wr1;
  const int q0 = u * 2, q1 = u * 2 + 1;
  const int k0c = q0 >> 3, n00 = (q0 & 7) * 8;
  const int k1c = q1 >> 3, n10 = (q1 & 7) * 8;

  auto tdm_x = [&](int s, int buf) {
    tdm_load_2d_bf16((unsigned)(uintptr_t)&Xs[buf][0][0],
                     Xb + (size_t)M0 * H_ + s * 32,
                     /*tile_x=*/32, /*tile_rows=*/128,
                     /*td0=*/H_, /*td1=*/T_, /*stride=*/H_);
  };
  auto wload = [&](int s) {
    int k0 = s * 32;
    wr0 = *(const uint4*)(Wsrc + (size_t)(k0 + k0c) * I_ + N0 + n00);
    wr1 = *(const uint4*)(Wsrc + (size_t)(k0 + k1c) * I_ + N0 + n10);
  };
  auto wstore = [&](int buf) {
    bf16_t* wb = (tid < 128) ? &Gs[buf][0][0] : &Vs[buf][0][0];
    const bf16_t* p0 = (const bf16_t*)&wr0;
    const bf16_t* p1 = (const bf16_t*)&wr1;
#pragma unroll
    for (int j = 0; j < 8; ++j) wb[(n00 + j) * PAD + k0c] = p0[j];
#pragma unroll
    for (int j = 0; j < 8; ++j) wb[(n10 + j) * PAD + k1c] = p1[j];
  };
  auto compute = [&](int buf) {
    v16bf a0 = ldsA(&Xs[buf][mrow + lr][0], klo);
    v16bf a1 = ldsA(&Xs[buf][mrow + 16 + lr][0], klo);
    v16bf g0 = ldsB(&Gs[buf][ncol + lr][kb]);
    v16bf g1 = ldsB(&Gs[buf][ncol + 16 + lr][kb]);
    v16bf v0 = ldsB(&Vs[buf][ncol + lr][kb]);
    v16bf v1 = ldsB(&Vs[buf][ncol + 16 + lr][kb]);
    accg[0][0] = wmma_bf16(a0, g0, accg[0][0]);
    accg[0][1] = wmma_bf16(a0, g1, accg[0][1]);
    accg[1][0] = wmma_bf16(a1, g0, accg[1][0]);
    accg[1][1] = wmma_bf16(a1, g1, accg[1][1]);
    accv[0][0] = wmma_bf16(a0, v0, accv[0][0]);
    accv[0][1] = wmma_bf16(a0, v1, accv[0][1]);
    accv[1][0] = wmma_bf16(a1, v0, accv[1][0]);
    accv[1][1] = wmma_bf16(a1, v1, accv[1][1]);
  };

  const int STEPS = H_ / 32;
  if (wid == 0) tdm_x(0, 0);
  wload(0);
  wstore(0);
  if (wid == 0) __builtin_amdgcn_s_wait_tensorcnt(0);
  __syncthreads();
#pragma unroll 1
  for (int s = 0; s < STEPS; ++s) {
    int buf = s & 1;
    if (s + 1 < STEPS) {
      if (wid == 0) tdm_x(s + 1, buf ^ 1);
      if (s + 2 < STEPS)
        __builtin_prefetch(Wsrc + (size_t)((s + 2) * 32 + k0c) * I_ + N0 + n00, 0, 0);
      wload(s + 1);
    }
    compute(buf);
    if (s + 1 < STEPS) {
      wstore(buf ^ 1);
      if (wid == 0) __builtin_amdgcn_s_wait_tensorcnt(0);
      __syncthreads();
    }
  }

  // Epilogue: A' = combine * gelu(g) * v, bf16
#pragma unroll
  for (int mi = 0; mi < 2; ++mi) {
    int rbase = M0 + mrow + mi * 16 + ((lane >> 4) << 3);
    float sc[8];
#pragma unroll
    for (int r = 0; r < 8; ++r) sc[r] = comb[(rbase + r) * E_ + e];
#pragma unroll
    for (int ni = 0; ni < 2; ++ni) {
      int col = N0 + ncol + ni * 16 + lr;
#pragma unroll
      for (int r = 0; r < 8; ++r) {
        float g = accg[mi][ni][r];
        float v = accv[mi][ni][r];
        float a = gelu_tanh(g) * v * sc[r];
        Ap[((size_t)e * T_ + rbase + r) * I_ + col] = (bf16_t)a;
      }
    }
  }
}

// ----------------------------------------------------------- output GEMM
// Y[T,H] = sum_e A'[e] @ Wout[e] -> single GEMM, K = E*I = 32768.
// A' tile via TDM; Wout tile via register transpose path.
__global__ void __launch_bounds__(256)
gemm_out(const bf16_t* __restrict__ Ap, const bf16_t* __restrict__ Woutb,
         float* __restrict__ out) {
  __shared__ __align__(16) bf16_t As[2][128][PAD];
  __shared__ __align__(16) bf16_t Ws[2][64][PAD];  // Wout tile, [n][k]

  const int M0 = blockIdx.y * 128;
  const int N0 = blockIdx.x * 64;

  const int tid  = threadIdx.x;
  const int wid  = tid >> 5, lane = tid & 31;
  const int wm   = wid >> 1, wn = wid & 1;
  const int mrow = wm * 32, ncol = wn * 32;
  const int lr   = lane & 15;
  const int klo  = (lane < 16) ? 0 : 8;
  const int kb   = (lane < 16) ? 0 : 8;

  const int k1 = tid >> 3, n1 = (tid & 7) * 8;

  const v8f_t zf = {0.f, 0.f, 0.f, 0.f, 0.f, 0.f, 0.f, 0.f};
  v8f_t acc[2][2];
#pragma unroll
  for (int i = 0; i < 2; ++i)
#pragma unroll
    for (int j = 0; j < 2; ++j) acc[i][j] = zf;

  uint4 wr0;
  auto tdm_a = [&](int s, int buf) {
    int e  = s >> 7;            // 128 K-steps per expert (I/32)
    int i0 = (s & 127) * 32;
    tdm_load_2d_bf16((unsigned)(uintptr_t)&As[buf][0][0],
                     Ap + ((size_t)e * T_ + M0) * I_ + i0,
                     /*tile_x=*/32, /*tile_rows=*/128,
                     /*td0=*/I_, /*td1=*/T_, /*stride=*/I_);
  };
  auto wload = [&](int s) {
    int e  = s >> 7;
    int i0 = (s & 127) * 32;
    wr0 = *(const uint4*)(Woutb + ((size_t)e * I_ + i0 + k1) * H_ + N0 + n1);
  };
  auto wstore = [&](int buf) {
    const bf16_t* p = (const bf16_t*)&wr0;
#pragma unroll
    for (int j = 0; j < 8; ++j) Ws[buf][n1 + j][k1] = p[j];
  };
  auto compute = [&](int buf) {
    v16bf a0 = ldsA(&As[buf][mrow + lr][0], klo);
    v16bf a1 = ldsA(&As[buf][mrow + 16 + lr][0], klo);
    v16bf b0 = ldsB(&Ws[buf][ncol + lr][kb]);
    v16bf b1 = ldsB(&Ws[buf][ncol + 16 + lr][kb]);
    acc[0][0] = wmma_bf16(a0, b0, acc[0][0]);
    acc[0][1] = wmma_bf16(a0, b1, acc[0][1]);
    acc[1][0] = wmma_bf16(a1, b0, acc[1][0]);
    acc[1][1] = wmma_bf16(a1, b1, acc[1][1]);
  };

  const int STEPS = E_ * (I_ / 32);  // 1024
  if (wid == 0) tdm_a(0, 0);
  wload(0);
  wstore(0);
  if (wid == 0) __builtin_amdgcn_s_wait_tensorcnt(0);
  __syncthreads();
#pragma unroll 1
  for (int s = 0; s < STEPS; ++s) {
    int buf = s & 1;
    if (s + 1 < STEPS) {
      if (wid == 0) tdm_a(s + 1, buf ^ 1);
      wload(s + 1);
    }
    compute(buf);
    if (s + 1 < STEPS) {
      wstore(buf ^ 1);
      if (wid == 0) __builtin_amdgcn_s_wait_tensorcnt(0);
      __syncthreads();
    }
  }

#pragma unroll
  for (int mi = 0; mi < 2; ++mi) {
    int rbase = M0 + mrow + mi * 16 + ((lane >> 4) << 3);
#pragma unroll
    for (int ni = 0; ni < 2; ++ni) {
      int col = N0 + ncol + ni * 16 + lr;
#pragma unroll
      for (int r = 0; r < 8; ++r)
        out[(size_t)(rbase + r) * H_ + col] = acc[mi][ni][r];
    }
  }
}

// ---------------------------------------------------------------- launch
extern "C" void kernel_launch(void* const* d_in, const int* in_sizes, int n_in,
                              void* d_out, int out_size, void* d_ws,
                              size_t ws_size, hipStream_t stream) {
  (void)in_sizes; (void)n_in; (void)out_size; (void)ws_size;
  const float* x    = (const float*)d_in[0];
  const float* Wg   = (const float*)d_in[1];
  const float* Win  = (const float*)d_in[2];
  const float* Wv   = (const float*)d_in[3];
  const float* Wout = (const float*)d_in[4];
  float* out = (float*)d_out;

  char* ws = (char*)d_ws;
  const size_t XB_OFF   = 0;
  const size_t COMB_OFF = XB_OFF   + (size_t)T_ * H_ * 2;
  const size_t WIN_OFF  = COMB_OFF + (size_t)T_ * E_ * 4;
  const size_t WV_OFF   = WIN_OFF  + (size_t)E_ * H_ * I_ * 2;
  const size_t WOUT_OFF = WV_OFF   + (size_t)E_ * H_ * I_ * 2;
  const size_t AP_OFF   = WOUT_OFF + (size_t)E_ * I_ * H_ * 2;

  bf16_t* Xb    = (bf16_t*)(ws + XB_OFF);
  float*  comb  = (float*)(ws + COMB_OFF);
  bf16_t* Winb  = (bf16_t*)(ws + WIN_OFF);
  bf16_t* Wvb   = (bf16_t*)(ws + WV_OFF);
  bf16_t* Woutb = (bf16_t*)(ws + WOUT_OFF);
  bf16_t* Ap    = (bf16_t*)(ws + AP_OFF);

  cvt_f32_bf16<<<1024, 256, 0, stream>>>(x, Xb, (size_t)T_ * H_ / 4);
  cvt_f32_bf16<<<4096, 256, 0, stream>>>(Win, Winb, (size_t)E_ * H_ * I_ / 4);
  cvt_f32_bf16<<<4096, 256, 0, stream>>>(Wv, Wvb, (size_t)E_ * H_ * I_ / 4);
  cvt_f32_bf16<<<4096, 256, 0, stream>>>(Wout, Woutb, (size_t)E_ * I_ * H_ / 4);

  router_kernel<<<T_ / 8, 256, 0, stream>>>(x, Wg, out + (size_t)T_ * H_, comb);

  gemm_gluact<<<dim3(I_ / 64, T_ / 128, E_), 256, 0, stream>>>(
      Xb, Winb, Wvb, comb, Ap);
  gemm_out<<<dim3(H_ / 64, T_ / 128), 256, 0, stream>>>(Ap, Woutb, out);
}
